// Gemma3Synapse270M_9251359555848
// MI455X (gfx1250) — matrix operand order repaired and checked by reading, not compiled, not run on hardware
//
#include <hip/hip_runtime.h>
#include <hip/hip_bf16.h>
#include <math.h>

// ---------------- model constants ----------------
constexpr int D   = 640;
constexpr int V   = 262208;
constexpr int L   = 18;
constexpr int H   = 4;
constexpr int HD  = 256;
constexpr int FFN = 2048;
constexpr int S   = 1024;
constexpr float ATTN_SCALE = 0.0625f;
constexpr float EMB_SCALE  = 25.298221281347036f;
constexpr int WIN_LOCAL   = 512;
constexpr int WIN_LONG    = 32768;
constexpr int ROPE_PERIOD = 6;
constexpr int VCHUNK = 15424;              // 241*64 ; V = 17 * VCHUNK
constexpr int NCHUNKS = V / VCHUNK;        // 17

typedef __bf16 bf16_t;
typedef __attribute__((ext_vector_type(16))) __bf16 v16bf;
typedef __attribute__((ext_vector_type(8)))  float  v8f;

#if __has_builtin(__builtin_amdgcn_global_load_async_to_lds_b32) && \
    __has_builtin(__builtin_amdgcn_s_wait_asynccnt)
#define USE_ASYNC_LDS 1
typedef __attribute__((address_space(1))) int gint_t;  // global
typedef __attribute__((address_space(3))) int lint_t;  // LDS
#endif

// ---------------------------------------------------------------------------
// WMMA GEMM:  C[M,N] (f32) = A[M,K] (f32) * B[K,N] (bf16, K-major)
// A converted to bf16 in-register; f32 accumulate via v_wmma_f32_16x16x32_bf16.
// Block = 256 threads = 8 waves; wave -> 16 rows x 64 cols of C.
// 2x-unrolled double-buffered pipeline: two fragment sets alternate, so the
// loop body is branch-free and needs no register rotation moves; each WMMA
// group overlaps with the other buffer's global_load_b128s.
// Requires K % 64 == 0 (all K here: 256/640/1024/2048).
// Batched via blockIdx.z (attention heads) with element strides.
// ---------------------------------------------------------------------------
__global__ __launch_bounds__(256) void gemm_wmma_kernel(
    const float* __restrict__ A, const bf16_t* __restrict__ B,
    float* __restrict__ C, int M, int N, int K, int lda, int ldb, int ldc,
    long strideA, long strideB, long strideC) {
  const float*  Ab = A + (long)blockIdx.z * strideA;
  const bf16_t* Bb = B + (long)blockIdx.z * strideB;
  float*        Cb = C + (long)blockIdx.z * strideC;

  const int wave = threadIdx.x >> 5;
  const int lane = threadIdx.x & 31;
  const int m0 = blockIdx.y * 128 + wave * 16;
  const int n0 = blockIdx.x * 64;
  if (m0 >= M) return;

  const int lrow = lane & 15;       // row within 16-row tile (A/C)
  const int hi   = (lane >> 4) & 1; // upper/lower half of wave
  const int arow = m0 + lrow;

  // ---- fragment loaders ----
  auto loadA = [&](int k0) -> v16bf {
    // 16-bit A 16x32 layout (ISA 7.12.2): lanes 0-15 hold K {0..7,16..23},
    // lanes 16-31 hold K {8..15,24..31} of the 32-wide K tile.
    const float* ap = Ab + (long)arow * lda + k0 + hi * 8;
    const float4 a0 = *(const float4*)(ap + 0);
    const float4 a1 = *(const float4*)(ap + 4);
    const float4 a2 = *(const float4*)(ap + 16);
    const float4 a3 = *(const float4*)(ap + 20);
    v16bf af;
    af[0]  = (bf16_t)a0.x; af[1]  = (bf16_t)a0.y; af[2]  = (bf16_t)a0.z; af[3]  = (bf16_t)a0.w;
    af[4]  = (bf16_t)a1.x; af[5]  = (bf16_t)a1.y; af[6]  = (bf16_t)a1.z; af[7]  = (bf16_t)a1.w;
    af[8]  = (bf16_t)a2.x; af[9]  = (bf16_t)a2.y; af[10] = (bf16_t)a2.z; af[11] = (bf16_t)a2.w;
    af[12] = (bf16_t)a3.x; af[13] = (bf16_t)a3.y; af[14] = (bf16_t)a3.z; af[15] = (bf16_t)a3.w;
    return af;
  };
  auto loadB = [&](int k0, int t) -> v16bf {
    // B 32x16 layout: lane holds column N = lrow, K {0..7,16..23} (+8 for hi).
    const bf16_t* bp = Bb + (long)(k0 + hi * 8) * ldb + n0 + t * 16 + lrow;
    v16bf bf;
#pragma unroll
    for (int i = 0; i < 8; ++i) {
      bf[i]     = bp[(long)i * ldb];
      bf[8 + i] = bp[(long)(16 + i) * ldb];
    }
    return bf;
  };

  v8f acc[4];
#pragma unroll
  for (int t = 0; t < 4; ++t)
#pragma unroll
    for (int r = 0; r < 8; ++r) acc[t][r] = 0.0f;

  const int KS = K >> 5;  // number of 32-wide K steps (always even here)

  // ---- prologue: buffer A <- step 0 ----
  v16bf afA = loadA(0);
  v16bf bfA[4];
#pragma unroll
  for (int t = 0; t < 4; ++t) bfA[t] = loadB(0, t);

  // ---- branch-free double-buffered main loop (2 K-steps per trip) ----
  int s = 0;
  for (; s + 2 < KS; s += 2) {
    // buffer B <- step s+1 (overlaps with WMMAs on buffer A)
    v16bf afB = loadA((s + 1) << 5);
    v16bf bfB[4];
#pragma unroll
    for (int t = 0; t < 4; ++t) bfB[t] = loadB((s + 1) << 5, t);
    if (s + 3 < KS)  // stream B ahead (global_prefetch_b8)
      __builtin_prefetch(
          (const void*)(Bb + (long)(((s + 3) << 5) + lrow + hi * 16) * ldb + n0), 0, 3);

#pragma unroll
    for (int t = 0; t < 4; ++t)
      acc[t] = __builtin_amdgcn_wmma_f32_16x16x32_bf16(
          false, afA, false, bfA[t], (short)0, acc[t], false, false);

    // buffer A <- step s+2 (overlaps with WMMAs on buffer B)
    afA = loadA((s + 2) << 5);
#pragma unroll
    for (int t = 0; t < 4; ++t) bfA[t] = loadB((s + 2) << 5, t);

#pragma unroll
    for (int t = 0; t < 4; ++t)
      acc[t] = __builtin_amdgcn_wmma_f32_16x16x32_bf16(
          false, afB, false, bfB[t], (short)0, acc[t], false, false);
  }

  // ---- epilogue: steps s (in buffer A) and s+1 remain ----
  {
    v16bf afB = loadA((s + 1) << 5);
    v16bf bfB[4];
#pragma unroll
    for (int t = 0; t < 4; ++t) bfB[t] = loadB((s + 1) << 5, t);

#pragma unroll
    for (int t = 0; t < 4; ++t)
      acc[t] = __builtin_amdgcn_wmma_f32_16x16x32_bf16(
          false, afA, false, bfA[t], (short)0, acc[t], false, false);
#pragma unroll
    for (int t = 0; t < 4; ++t)
      acc[t] = __builtin_amdgcn_wmma_f32_16x16x32_bf16(
          false, afB, false, bfB[t], (short)0, acc[t], false, false);
  }

  // ---- store C (8 VGPRs; lanes 16-31 hold M+8) ----
  const int crow = m0 + hi * 8;
#pragma unroll
  for (int t = 0; t < 4; ++t) {
    const int col = n0 + t * 16 + lrow;
#pragma unroll
    for (int r = 0; r < 8; ++r)
      Cb[(long)(crow + r) * ldc + col] = acc[t][r];
  }
}

// ---------------------------------------------------------------------------
// Tiled transpose + f32->bf16:  W[Nrows][Kcols] f32  ->  Wt[Kcols][Nrows] bf16
// Coalesced loads (along K) and stores (along N); 32x32 tile in LDS.
// Global->LDS fill uses CDNA5 async-to-LDS (ASYNCcnt) when available.
// ---------------------------------------------------------------------------
__global__ __launch_bounds__(256) void transpose_to_bf16_kernel(
    const float* __restrict__ W, bf16_t* __restrict__ Wt, int Nrows, int Kcols) {
  __shared__ float tile[32][33];
  const int k0 = blockIdx.x * 32;
  const int n0 = blockIdx.y * 32;
  const int tx = threadIdx.x & 31;
  const int ty = threadIdx.x >> 5;  // 0..7

#ifdef USE_ASYNC_LDS
#pragma unroll
  for (int i = 0; i < 32; i += 8) {
    const float* gp = &W[(long)(n0 + ty + i) * Kcols + k0 + tx];
    __builtin_amdgcn_global_load_async_to_lds_b32(
        (gint_t*)gp, (lint_t*)&tile[ty + i][tx], 0, 0);
  }
  __builtin_amdgcn_s_wait_asynccnt(0);
  __syncthreads();
#else
#pragma unroll
  for (int i = 0; i < 32; i += 8)
    tile[ty + i][tx] = W[(long)(n0 + ty + i) * Kcols + k0 + tx];
  __syncthreads();
#endif

#pragma unroll
  for (int i = 0; i < 32; i += 8)
    Wt[(long)(k0 + ty + i) * Nrows + n0 + tx] = (bf16_t)tile[tx][ty + i];
}

// ---------------- plain f32 -> bf16 convert ----------------
__global__ void convert_bf16_kernel(const float* __restrict__ in,
                                    bf16_t* __restrict__ out, long n) {
  long i = (long)blockIdx.x * blockDim.x + threadIdx.x;
  if (i < n) out[i] = (bf16_t)in[i];
}

// ---------------- embedding gather + scale ----------------
__global__ void embed_kernel(const int* __restrict__ ids,
                             const float* __restrict__ embed,
                             float* __restrict__ x) {
  long i = (long)blockIdx.x * blockDim.x + threadIdx.x;
  if (i >= (long)S * D) return;
  int s = (int)(i / D), d = (int)(i % D);
  x[i] = embed[(long)ids[s] * D + d] * EMB_SCALE;
}

// ---------------- RMSNorm: out = x*rsqrt(mean(x^2)+eps)*(1+w) --------------
__global__ __launch_bounds__(256) void rmsnorm_kernel(
    const float* __restrict__ in, const float* __restrict__ w,
    float* __restrict__ out, int cols, int addmode) {
  const long row = blockIdx.x;
  const float* xr = in + row * cols;
  __shared__ float red[256];

  float vals[4];
  int nv = 0;
  float ss = 0.0f;
  for (int c = threadIdx.x; c < cols; c += 256) {
    float v = xr[c];
    vals[nv++] = v;
    ss += v * v;
  }
  red[threadIdx.x] = ss;
  __syncthreads();
  for (int st = 128; st > 0; st >>= 1) {
    if (threadIdx.x < st) red[threadIdx.x] += red[threadIdx.x + st];
    __syncthreads();
  }
  const float rs = rsqrtf(red[0] / (float)cols + 1e-6f);

  nv = 0;
  for (int c = threadIdx.x; c < cols; c += 256) {
    float r = vals[nv++] * rs * (1.0f + w[c]);
    if (addmode) out[row * cols + c] += r;
    else         out[row * cols + c] = r;
  }
}

// ---------------- RoPE (in place) ----------------
__global__ void rope_kernel(float* __restrict__ x, int heads, float theta) {
  const int half = HD / 2;
  long idx = (long)blockIdx.x * blockDim.x + threadIdx.x;
  long total = (long)S * heads * half;
  if (idx >= total) return;
  int j = (int)(idx % half);
  long sh = idx / half;
  int h = (int)(sh % heads);
  long s = sh / heads;
  float inv = powf(theta, -(float)j / (float)half);
  float f = (float)s * inv;
  float c = cosf(f), sn = sinf(f);
  float* p = x + (s * heads + h) * (long)HD;
  float x1 = p[j], x2 = p[j + half];
  p[j]        = x1 * c - x2 * sn;
  p[j + half] = x2 * c + x1 * sn;
}

// ---------------- masked softmax over score rows (in place) ----------------
__global__ __launch_bounds__(256) void softmax_kernel(float* __restrict__ scores,
                                                      int win) {
  const int qpos = blockIdx.x;
  float* row = scores + ((long)blockIdx.y * S + qpos) * (long)S;
  __shared__ float red[256];

  float v[4];
  float mx = -3.0e38f;
#pragma unroll
  for (int t = 0; t < 4; ++t) {
    int kpos = threadIdx.x + t * 256;
    int rel = qpos - kpos;
    bool ok = (rel >= 0) && (rel < win);
    float sv = ok ? row[kpos] * ATTN_SCALE : -1.0e30f;
    v[t] = sv;
    mx = fmaxf(mx, sv);
  }
  red[threadIdx.x] = mx;
  __syncthreads();
  for (int st = 128; st > 0; st >>= 1) {
    if (threadIdx.x < st) red[threadIdx.x] = fmaxf(red[threadIdx.x], red[threadIdx.x + st]);
    __syncthreads();
  }
  mx = red[0];
  __syncthreads();

  float sum = 0.0f;
#pragma unroll
  for (int t = 0; t < 4; ++t) {
    v[t] = __expf(v[t] - mx);
    sum += v[t];
  }
  red[threadIdx.x] = sum;
  __syncthreads();
  for (int st = 128; st > 0; st >>= 1) {
    if (threadIdx.x < st) red[threadIdx.x] += red[threadIdx.x + st];
    __syncthreads();
  }
  const float inv = 1.0f / red[0];
#pragma unroll
  for (int t = 0; t < 4; ++t) row[threadIdx.x + t * 256] = v[t] * inv;
}

// ---------------- gelu(tanh)(g) * u -> g ----------------
__global__ void gelu_mul_kernel(float* __restrict__ g, const float* __restrict__ u,
                                long n) {
  long i = (long)blockIdx.x * blockDim.x + threadIdx.x;
  if (i >= n) return;
  float x = g[i];
  float t = tanhf(0.7978845608028654f * (x + 0.044715f * x * x * x));
  g[i] = 0.5f * x * (1.0f + t) * u[i];
}

// ---------------------------------------------------------------------------
// host helpers
// ---------------------------------------------------------------------------
static inline void gemm(const float* A, const bf16_t* B, float* C, int M, int N,
                        int K, int lda, int ldb, int ldc, int batch, long sA,
                        long sB, long sC, hipStream_t st) {
  dim3 grid(N / 64, (M + 127) / 128, batch);
  gemm_wmma_kernel<<<grid, 256, 0, st>>>(A, B, C, M, N, K, lda, ldb, ldc, sA, sB, sC);
}

// stage W[Nrows][Kcols] f32 as Wt[Kcols][Nrows] bf16
static inline void stageT(const float* W, bf16_t* Wt, int Nrows, int Kcols,
                          hipStream_t st) {
  dim3 grid(Kcols / 32, Nrows / 32);
  transpose_to_bf16_kernel<<<grid, 256, 0, st>>>(W, Wt, Nrows, Kcols);
}

extern "C" void kernel_launch(void* const* d_in, const int* in_sizes, int n_in,
                              void* d_out, int out_size, void* d_ws,
                              size_t ws_size, hipStream_t stream) {
  (void)in_sizes; (void)n_in; (void)out_size; (void)ws_size;

  const int*   input_ids    = (const int*)d_in[0];
  const float* embed        = (const float*)d_in[1];
  const float* q_proj       = (const float*)d_in[2];
  const float* k_proj       = (const float*)d_in[3];
  const float* v_proj       = (const float*)d_in[4];
  const float* o_proj       = (const float*)d_in[5];
  const float* q_norm       = (const float*)d_in[6];
  const float* k_norm       = (const float*)d_in[7];
  const float* in_ln        = (const float*)d_in[8];
  const float* post_attn_ln = (const float*)d_in[9];
  const float* pre_ffn_ln   = (const float*)d_in[10];
  const float* post_ffn_ln  = (const float*)d_in[11];
  const float* gate_proj    = (const float*)d_in[12];
  const float* up_proj      = (const float*)d_in[13];
  const float* down_proj    = (const float*)d_in[14];
  const float* final_norm   = (const float*)d_in[15];

  // ---- workspace layout: f32 activations then bf16 staging (~75 MB) ----
  float* ws = (float*)d_ws;
  float* x  = ws;                       // [S][D]
  float* xn = x  + (long)S * D;         // [S][D]
  float* q  = xn + (long)S * D;         // [S][H*HD]
  float* kb = q  + (long)S * H * HD;    // [S][HD]
  float* vb = kb + (long)S * HD;        // [S][HD]
  float* sc = vb + (long)S * HD;        // [H][S][S]
  float* ao = sc + (long)H * S * S;     // [S][H*HD]
  float* ap = ao + (long)S * H * HD;    // [S][D]
  float* g  = ap + (long)S * D;         // [S][FFN]
  float* u  = g  + (long)S * FFN;       // [S][FFN]
  float* m  = u  + (long)S * FFN;       // [S][D]
  bf16_t* wt  = (bf16_t*)(m + (long)S * D);      // [>= D*VCHUNK] staging
  bf16_t* kbt = wt  + (long)D * VCHUNK;          // [HD][S]
  bf16_t* vbt = kbt + (long)HD * S;              // [S][HD]

  // ---- embedding ----
  {
    long n = (long)S * D;
    embed_kernel<<<(int)((n + 255) / 256), 256, 0, stream>>>(input_ids, embed, x);
  }

  for (int i = 0; i < L; ++i) {
    const bool  is_global = ((i + 1) % ROPE_PERIOD) == 0;
    const float theta = is_global ? 1000000.0f : 10000.0f;
    const int   win   = is_global ? WIN_LONG : WIN_LOCAL;

    const float* qp  = q_proj    + (long)i * H * HD * D;
    const float* kp  = k_proj    + (long)i * HD * D;
    const float* vp  = v_proj    + (long)i * HD * D;
    const float* op  = o_proj    + (long)i * D * H * HD;
    const float* qn  = q_norm    + (long)i * HD;
    const float* kn  = k_norm    + (long)i * HD;
    const float* iln = in_ln     + (long)i * D;
    const float* pal = post_attn_ln + (long)i * D;
    const float* pfl = pre_ffn_ln   + (long)i * D;
    const float* pol = post_ffn_ln  + (long)i * D;
    const float* gp  = gate_proj + (long)i * FFN * D;
    const float* up  = up_proj   + (long)i * FFN * D;
    const float* dp  = down_proj + (long)i * D * FFN;

    // xn = rms(x, in_ln)
    rmsnorm_kernel<<<S, 256, 0, stream>>>(x, iln, xn, D, 0);

    // q/k/v projections: stage W^T as bf16 [D][Nout], then GEMM
    stageT(qp, wt, H * HD, D, stream);
    gemm(xn, wt, q,  S, H * HD, D, D, H * HD, H * HD, 1, 0, 0, 0, stream);
    stageT(kp, wt, HD, D, stream);
    gemm(xn, wt, kb, S, HD, D, D, HD, HD, 1, 0, 0, 0, stream);
    stageT(vp, wt, HD, D, stream);
    gemm(xn, wt, vb, S, HD, D, D, HD, HD, 1, 0, 0, 0, stream);

    // per-head q-norm (rows = S*H of length HD), k-norm
    rmsnorm_kernel<<<S * H, 256, 0, stream>>>(q,  qn, q,  HD, 0);
    rmsnorm_kernel<<<S,     256, 0, stream>>>(kb, kn, kb, HD, 0);

    // RoPE (in place)
    {
      long nq = (long)S * H * (HD / 2);
      rope_kernel<<<(int)((nq + 255) / 256), 256, 0, stream>>>(q, H, theta);
      long nk = (long)S * (HD / 2);
      rope_kernel<<<(int)((nk + 255) / 256), 256, 0, stream>>>(kb, 1, theta);
    }

    // scores[h] = Q_h * K^T : stage K^T bf16 [HD][S], batch over heads
    stageT(kb, kbt, S, HD, stream);
    gemm(q, kbt, sc, S, S, HD, H * HD, S, S, H, (long)HD, 0, (long)S * S, stream);

    // masked softmax (scale + window/causal mask folded in)
    softmax_kernel<<<dim3(S, H), 256, 0, stream>>>(sc, win);

    // attn[h] = P_h * V : V already K-major, just convert to bf16
    {
      long n = (long)S * HD;
      convert_bf16_kernel<<<(int)((n + 255) / 256), 256, 0, stream>>>(vb, vbt, n);
    }
    gemm(sc, vbt, ao, S, HD, S, S, HD, H * HD, H, (long)S * S, 0, (long)HD, stream);

    // o-projection, residual with post-attn norm
    stageT(op, wt, D, H * HD, stream);   // [640][1024] -> [1024][640]
    gemm(ao, wt, ap, S, D, H * HD, H * HD, D, D, 1, 0, 0, 0, stream);
    rmsnorm_kernel<<<S, 256, 0, stream>>>(ap, pal, x, D, 1);

    // FFN
    rmsnorm_kernel<<<S, 256, 0, stream>>>(x, pfl, xn, D, 0);
    stageT(gp, wt, FFN, D, stream);
    gemm(xn, wt, g, S, FFN, D, D, FFN, FFN, 1, 0, 0, 0, stream);
    stageT(up, wt, FFN, D, stream);
    gemm(xn, wt, u, S, FFN, D, D, FFN, FFN, 1, 0, 0, 0, stream);
    {
      long n = (long)S * FFN;
      gelu_mul_kernel<<<(int)((n + 255) / 256), 256, 0, stream>>>(g, u, n);
    }
    stageT(dp, wt, D, FFN, stream);      // [640][2048] -> [2048][640]
    gemm(g, wt, m, S, D, FFN, FFN, D, D, 1, 0, 0, 0, stream);
    rmsnorm_kernel<<<S, 256, 0, stream>>>(m, pol, x, D, 1);
  }

  // ---- final norm + logits = xn * embed^T, chunked over V ----
  rmsnorm_kernel<<<S, 256, 0, stream>>>(x, final_norm, xn, D, 0);
  float* logits = (float*)d_out;
  for (int c = 0; c < NCHUNKS; ++c) {
    const float* wchunk = embed + (long)c * VCHUNK * D;
    stageT(wchunk, wt, VCHUNK, D, stream);               // -> [D][VCHUNK] bf16
    gemm(xn, wt, logits + (long)c * VCHUNK, S, VCHUNK, D, D, VCHUNK, V,
         1, 0, 0, 0, stream);
  }
}
